// SelfAttention_33483565040242
// MI455X (gfx1250) — matrix operand order repaired and checked by reading, not compiled, not run on hardware
//
#include <hip/hip_runtime.h>
#include <hip/hip_bf16.h>

#define B_   2
#define S_   2048
#define D_   2048
#define NH_  32
#define NKV_ 8
#define HD_  64
#define NREP_ (NH_/NKV_)

typedef __attribute__((ext_vector_type(16))) __bf16 v16bf;
typedef __attribute__((ext_vector_type(8)))  __bf16 v8bf;
typedef __attribute__((ext_vector_type(8)))  float  v8f;
typedef __attribute__((ext_vector_type(4)))  unsigned int v4u_t;
typedef __attribute__((ext_vector_type(8)))  int v8i_t;
typedef __attribute__((ext_vector_type(4)))  int v4i_t;

#if defined(__has_builtin)
#if __has_builtin(__builtin_amdgcn_tensor_load_to_lds) && \
    __has_builtin(__builtin_amdgcn_s_wait_tensorcnt)
#define USE_TDM 1
#endif
#endif

__device__ __forceinline__ v16bf cat8(v8bf lo, v8bf hi) {
  return __builtin_shufflevector(lo, hi, 0,1,2,3,4,5,6,7,8,9,10,11,12,13,14,15);
}

// A-operand (16x32 bf16): lane holds row (lane&15); element e -> K = 2*kb+e
// (contiguous); kb = (lane<16)?0:8.  Two contiguous 16B loads.
__device__ __forceinline__ v16bf load_a(const __bf16* rowbase, int kb) {
  v8bf lo = *(const v8bf*)(rowbase + 2 * kb);
  v8bf hi = *(const v8bf*)(rowbase + 2 * kb + 8);
  return cat8(lo, hi);
}
// B-operand (32x16 bf16): lane holds column (lane&15); K = kb16..kb16+15
// contiguous; kb16 = (lane<16)?0:16.
__device__ __forceinline__ v16bf load_b(const __bf16* colbase, int kb16) {
  v8bf lo = *(const v8bf*)(colbase + kb16);
  v8bf hi = *(const v8bf*)(colbase + kb16 + 8);
  return cat8(lo, hi);
}

__device__ __forceinline__ v8f wmma_bf16(v16bf a, v16bf b, v8f c) {
  return __builtin_amdgcn_wmma_f32_16x16x32_bf16(false, a, false, b, (short)0, c,
                                                 false, false);
}

// ---------------- elementwise conversion kernels ----------------
__global__ void cvt_f32_bf16(const float* __restrict__ x, __bf16* __restrict__ y, int n) {
  int i = blockIdx.x * blockDim.x + threadIdx.x;
  if (i < n) y[i] = (__bf16)x[i];
}

// W (K x N, row major f32) -> WT (N x K, row major bf16)
__global__ void transpose_cvt(const float* __restrict__ w, __bf16* __restrict__ wt,
                              int K, int N) {
  int i = blockIdx.x * blockDim.x + threadIdx.x;
  if (i < K * N) {
    int k = i / N, n = i % N;
    wt[(size_t)n * K + k] = (__bf16)w[i];
  }
}

// RoPE on (B,S,H,HD) f32 -> bf16, pair index p in [0,32)
__global__ void rope_cvt(const float* __restrict__ src, __bf16* __restrict__ dst,
                         const float* __restrict__ cosf, const float* __restrict__ sinf,
                         int H, int n_pairs) {
  int i = blockIdx.x * blockDim.x + threadIdx.x;
  if (i >= n_pairs) return;
  int p = i & 31;                   // HD/2 == 32
  int s = (i / (32 * H)) % S_;
  float x0 = src[2 * i], x1 = src[2 * i + 1];
  float c = cosf[s * 32 + p], sn = sinf[s * 32 + p];
  dst[2 * i]     = (__bf16)(x0 * c - x1 * sn);
  dst[2 * i + 1] = (__bf16)(x0 * sn + x1 * c);
}

// ------- WMMA GEMM: C(MxN,f32) = A(MxK,bf16) x WT(NxK,bf16)^T, 32x64/wave ----
__global__ __launch_bounds__(32) void gemm_bf16_wmma(
    const __bf16* __restrict__ A, const __bf16* __restrict__ WT,
    float* __restrict__ C, int M, int N, int K) {
  const int lane = threadIdx.x;
  const int half = (lane < 16) ? 0 : 1;
  const int lr   = lane & 15;
  const int row0 = blockIdx.y * 32;
  const int col0 = blockIdx.x * 64;

  const __bf16* arow0 = A + (size_t)(row0 + lr) * K;
  const __bf16* arow1 = A + (size_t)(row0 + 16 + lr) * K;
  const __bf16* wcol0 = WT + (size_t)(col0 +  0 + lr) * K;
  const __bf16* wcol1 = WT + (size_t)(col0 + 16 + lr) * K;
  const __bf16* wcol2 = WT + (size_t)(col0 + 32 + lr) * K;
  const __bf16* wcol3 = WT + (size_t)(col0 + 48 + lr) * K;
  const int kbA = half * 8;
  const int kbB = half * 16;

  v8f p00 = {}, p01 = {}, p02 = {}, p03 = {};
  v8f p10 = {}, p11 = {}, p12 = {}, p13 = {};
  for (int k0 = 0; k0 < K; k0 += 32) {
    v16bf a0 = load_a(arow0 + k0, kbA);
    v16bf a1 = load_a(arow1 + k0, kbA);
    v16bf b0 = load_b(wcol0 + k0, kbB);
    p00 = wmma_bf16(a0, b0, p00);
    p10 = wmma_bf16(a1, b0, p10);
    v16bf b1 = load_b(wcol1 + k0, kbB);
    p01 = wmma_bf16(a0, b1, p01);
    p11 = wmma_bf16(a1, b1, p11);
    v16bf b2 = load_b(wcol2 + k0, kbB);
    p02 = wmma_bf16(a0, b2, p02);
    p12 = wmma_bf16(a1, b2, p12);
    v16bf b3 = load_b(wcol3 + k0, kbB);
    p03 = wmma_bf16(a0, b3, p03);
    p13 = wmma_bf16(a1, b3, p13);
  }
  v8f acc0[4] = {p00, p01, p02, p03};
  v8f acc1[4] = {p10, p11, p12, p13};
  #pragma unroll
  for (int t = 0; t < 4; ++t) {
    int n = col0 + t * 16 + lr;
    #pragma unroll
    for (int r = 0; r < 8; ++r) {
      int m0 = row0 + r + half * 8;
      C[(size_t)m0 * N + n] = acc0[t][r];
      C[(size_t)(m0 + 16) * N + n] = acc1[t][r];
    }
  }
}

// ---------------- flash attention, one wave per (b, h, 16-query tile) ---------
__global__ __launch_bounds__(32) void flash_attn_wmma(
    const __bf16* __restrict__ q, const __bf16* __restrict__ k,
    const __bf16* __restrict__ v, __bf16* __restrict__ o) {
  __shared__ __attribute__((aligned(16))) __bf16 vtile[2][32 * 64];

  const int lane = threadIdx.x;
  const int half = (lane < 16) ? 0 : 1;
  const int lr   = lane & 15;
  const int q0   = blockIdx.x * 16;
  const int h    = blockIdx.y;
  const int b    = blockIdx.z;
  const int kvh  = h / NREP_;
  const int kbA  = half * 8;
  const int kbB  = half * 16;

  // Q^T B-operand tiles (dims x queries), fixed for the block.
  const __bf16* qrow = q + (((size_t)(b * S_ + q0 + lr)) * NH_ + h) * HD_;
  v16bf bq0 = load_b(qrow,      kbB);   // head dims 0..31
  v16bf bq1 = load_b(qrow + 32, kbB);   // head dims 32..63

  v8f oacc0 = {}, oacc1 = {}, oacc2 = {}, oacc3 = {};
  float mrun = -__builtin_inff();
  float lrun = 0.0f;
  const int qidx = q0 + lr;             // this lane's query column

  const int ntiles = (q0 + 15) / 32 + 1;

#ifdef USE_TDM
  // Issue a 2D Tensor-Data-Mover load of a 32-key x 64-dim bf16 V tile.
  // D# per CDNA5 ISA ch.8: group0 = {count, lds_addr, global_addr, type=2};
  // group1 = {data_size=2B, tensor_dim0=64, tensor_dim1=32, tile_dim0=64,
  //           tile_dim1=32, tensor_dim0_stride=NKV*HD=512}.
  auto tdm_issue = [&](int tt) {
    const int ss0 = tt * 32;
    unsigned long long ga =
        (unsigned long long)(const void*)(v + (((size_t)(b * S_ + ss0)) * NKV_ + kvh) * HD_);
    unsigned lds_base = (unsigned)(size_t)(&vtile[tt & 1][0]);  // low 32 bits = LDS offset
    v4u_t g0;
    g0[0] = 1u;                                   // count=1, user descriptor
    g0[1] = lds_base;                             // lds_addr
    g0[2] = (unsigned)ga;                         // global_addr[31:0]
    g0[3] = ((unsigned)(ga >> 32) & 0x01FFFFFFu)  // global_addr[56:32]
            | (2u << 30);                         // type=2 ("image")
    v8i_t g1;
    g1[0] = (int)(1u << 16);                      // data_size=1 (2 bytes)
    g1[1] = (int)((unsigned)HD_ << 16);           // tensor_dim0[15:0]=64
    g1[2] = (int)(32u << 16);                     // dim0[31:16]=0 | tensor_dim1[15:0]=32
    g1[3] = (int)((unsigned)HD_ << 16);           // tensor_dim1[31:16]=0 | tile_dim0=64
    g1[4] = 32;                                   // tile_dim1=32, tile_dim2=0
    g1[5] = NKV_ * HD_;                           // tensor_dim0_stride[31:0]=512
    g1[6] = 0;                                    // stride[47:32]=0 | dim1_stride lo
    g1[7] = 0;
    v4i_t z4 = {0, 0, 0, 0};
#if __has_include(<hip/amd_detail/amd_gfx1250_TDM.h>)
    v8i_t z8 = {0, 0, 0, 0, 0, 0, 0, 0};
    __builtin_amdgcn_tensor_load_to_lds(g0, g1, z4, z4, z8, 0);
#else
    __builtin_amdgcn_tensor_load_to_lds(g0, g1, z4, z4, 0);
#endif
  };
  tdm_issue(0);
#endif

  for (int t = 0; t < ntiles; ++t) {
    const int s0 = t * 32;
    const __bf16* vt = &vtile[t & 1][0];

#ifdef USE_TDM
    if (t + 1 < ntiles) {
      tdm_issue(t + 1);                      // prefetch next tile (other buffer)
      __builtin_amdgcn_s_wait_tensorcnt(1);  // wait for current tile only
    } else {
      __builtin_amdgcn_s_wait_tensorcnt(0);
    }
#else
    // Fallback: stage V key-tile (32 keys x 64 dims): one key row per lane.
    __syncthreads();
    {
      const __bf16* vrow = v + (((size_t)(b * S_ + s0 + lane)) * NKV_ + kvh) * HD_;
      const uint4* src = (const uint4*)vrow;
      uint4* dst = (uint4*)&vtile[t & 1][lane * 64];
      #pragma unroll
      for (int j = 0; j < 8; ++j) dst[j] = src[j];
    }
    __syncthreads();
#endif

    // S^T = K x Q^T  (keys x queries); two 16-key halves, K-dim = HD (2 steps).
    const __bf16* krow0 = k + (((size_t)(b * S_ + s0      + lr)) * NKV_ + kvh) * HD_;
    const __bf16* krow1 = k + (((size_t)(b * S_ + s0 + 16 + lr)) * NKV_ + kvh) * HD_;
    v8f sc0 = {}, sc1 = {};
    sc0 = wmma_bf16(load_a(krow0,      kbA), bq0, sc0);
    sc0 = wmma_bf16(load_a(krow0 + 32, kbA), bq1, sc0);
    sc1 = wmma_bf16(load_a(krow1,      kbA), bq0, sc1);
    sc1 = wmma_bf16(load_a(krow1 + 32, kbA), bq1, sc1);

    // Scale, causal mask, online softmax stats (per-lane = per-query column).
    float p0[8], p1[8];
    float tmax = -__builtin_inff();
    #pragma unroll
    for (int r = 0; r < 8; ++r) {
      int key0 = s0 + r + half * 8;
      int key1 = key0 + 16;
      float x0 = (key0 <= qidx) ? sc0[r] * 0.125f : -__builtin_inff();
      float x1 = (key1 <= qidx) ? sc1[r] * 0.125f : -__builtin_inff();
      p0[r] = x0; p1[r] = x1;
      tmax = fmaxf(tmax, fmaxf(x0, x1));
    }
    tmax = fmaxf(tmax, __shfl_xor(tmax, 16, 32));   // combine row halves
    float mnew  = fmaxf(mrun, tmax);
    float alpha = __expf(mrun - mnew);
    float tsum  = 0.0f;
    #pragma unroll
    for (int r = 0; r < 8; ++r) {
      p0[r] = __expf(p0[r] - mnew);
      p1[r] = __expf(p1[r] - mnew);
      tsum += p0[r] + p1[r];
    }
    tsum += __shfl_xor(tsum, 16, 32);
    lrun = lrun * alpha + tsum;
    mrun = mnew;
    #pragma unroll
    for (int r = 0; r < 8; ++r) {
      oacc0[r] *= alpha; oacc1[r] *= alpha; oacc2[r] *= alpha; oacc3[r] *= alpha;
    }

    // Build B = P^T (32 keys x 16 queries) in B-operand layout via shuffles.
    v16bf bp;
    #pragma unroll
    for (int e = 0; e < 8; ++e) {
      float t0 = __shfl_xor(p0[e], 16, 32);
      float t1 = __shfl_xor(p1[e], 16, 32);
      bp[e]     = (__bf16)((half == 0) ? p0[e] : t1);
      bp[8 + e] = (__bf16)((half == 0) ? t0    : p1[e]);
    }

    // O^T += V^T x P^T : A = V^T (dims x keys) from LDS, 4 dim-tiles.
    #pragma unroll
    for (int t2 = 0; t2 < 4; ++t2) {
      const int dim = t2 * 16 + lr;   // A row = head dim
      v16bf av;
      #pragma unroll
      for (int e = 0; e < 8; ++e) {
        av[e]     = vt[(2 * kbA + e)     * 64 + dim];  // keys 2kb..2kb+7
        av[8 + e] = vt[(2 * kbA + 8 + e) * 64 + dim];  // keys 2kb+8..2kb+15
      }
      if (t2 == 0) oacc0 = wmma_bf16(av, bp, oacc0);
      if (t2 == 1) oacc1 = wmma_bf16(av, bp, oacc1);
      if (t2 == 2) oacc2 = wmma_bf16(av, bp, oacc2);
      if (t2 == 3) oacc3 = wmma_bf16(av, bp, oacc3);
    }
  }

  // Normalize and store O (b, s, h, hd) as bf16 for the output projection.
  float inv = 1.0f / lrun;
  __bf16* orow = o + (((size_t)(b * S_ + qidx)) * NH_ + h) * HD_;
  v8f oaccs[4] = {oacc0, oacc1, oacc2, oacc3};
  #pragma unroll
  for (int t2 = 0; t2 < 4; ++t2) {
    #pragma unroll
    for (int r = 0; r < 8; ++r) {
      orow[t2 * 16 + r + half * 8] = (__bf16)(oaccs[t2][r] * inv);
    }
  }
}

extern "C" void kernel_launch(void* const* d_in, const int* in_sizes, int n_in,
                              void* d_out, int out_size, void* d_ws, size_t ws_size,
                              hipStream_t stream) {
  (void)in_sizes; (void)n_in; (void)out_size; (void)ws_size;
  const float* x    = (const float*)d_in[0];
  const float* fcos = (const float*)d_in[1];
  const float* fsin = (const float*)d_in[2];
  const float* wq   = (const float*)d_in[3];
  const float* wk   = (const float*)d_in[4];
  const float* wv   = (const float*)d_in[5];
  const float* wo   = (const float*)d_in[6];
  float* out = (float*)d_out;

  const size_t BS  = (size_t)B_ * S_;
  const size_t DKV = (size_t)NKV_ * HD_;   // 512
  char* p = (char*)d_ws;
  auto alloc = [&](size_t bytes) -> void* {
    void* r = (void*)p;
    p += (bytes + 255) & ~(size_t)255;
    return r;
  };
  __bf16* xb  = (__bf16*)alloc(BS * D_ * 2);
  __bf16* wqT = (__bf16*)alloc((size_t)D_ * D_ * 2);
  __bf16* wkT = (__bf16*)alloc((size_t)D_ * DKV * 2);
  __bf16* wvT = (__bf16*)alloc((size_t)D_ * DKV * 2);
  __bf16* woT = (__bf16*)alloc((size_t)D_ * D_ * 2);
  float*  qf  = (float*) alloc(BS * D_ * 4);
  float*  kf  = (float*) alloc(BS * DKV * 4);
  float*  vf  = (float*) alloc(BS * DKV * 4);
  __bf16* qb  = (__bf16*)alloc(BS * D_ * 2);
  __bf16* kb  = (__bf16*)alloc(BS * DKV * 2);
  __bf16* vb  = (__bf16*)alloc(BS * DKV * 2);
  __bf16* ob  = (__bf16*)alloc(BS * D_ * 2);

  int n;
  n = (int)(BS * D_);
  cvt_f32_bf16<<<(n + 255) / 256, 256, 0, stream>>>(x, xb, n);
  n = D_ * D_;
  transpose_cvt<<<(n + 255) / 256, 256, 0, stream>>>(wq, wqT, D_, D_);
  n = (int)((size_t)D_ * DKV);
  transpose_cvt<<<(n + 255) / 256, 256, 0, stream>>>(wk, wkT, D_, (int)DKV);
  transpose_cvt<<<(n + 255) / 256, 256, 0, stream>>>(wv, wvT, D_, (int)DKV);
  n = D_ * D_;
  transpose_cvt<<<(n + 255) / 256, 256, 0, stream>>>(wo, woT, D_, D_);

  dim3 blk(32);
  dim3 gq(D_ / 64, (unsigned)(BS / 32));
  gemm_bf16_wmma<<<gq, blk, 0, stream>>>(xb, wqT, qf, (int)BS, D_, D_);
  dim3 gkv((unsigned)(DKV / 64), (unsigned)(BS / 32));
  gemm_bf16_wmma<<<gkv, blk, 0, stream>>>(xb, wkT, kf, (int)BS, (int)DKV, D_);
  gemm_bf16_wmma<<<gkv, blk, 0, stream>>>(xb, wvT, vf, (int)BS, (int)DKV, D_);

  n = (int)(BS * NH_ * 32);
  rope_cvt<<<(n + 255) / 256, 256, 0, stream>>>(qf, qb, fcos, fsin, NH_, n);
  n = (int)(BS * NKV_ * 32);
  rope_cvt<<<(n + 255) / 256, 256, 0, stream>>>(kf, kb, fcos, fsin, NKV_, n);
  n = (int)(BS * DKV);
  cvt_f32_bf16<<<(n + 255) / 256, 256, 0, stream>>>(vf, vb, n);

  dim3 ga(S_ / 16, NH_, B_);
  flash_attn_wmma<<<ga, blk, 0, stream>>>(qb, kb, vb, ob);

  dim3 go(D_ / 64, (unsigned)(BS / 32));
  gemm_bf16_wmma<<<go, blk, 0, stream>>>(ob, woT, out, (int)BS, D_, D_);
}
